// PerceiverAR_75488345194710
// MI455X (gfx1250) — compile-verified, matrix operand order checked by reading, not run on hardware
//
#include <hip/hip_runtime.h>
#include <hip/hip_bf16.h>

// ---------------------------------------------------------------------------
// PerceiverAR forward for MI455X (gfx1250, wave32, WMMA bf16 16x16x32)
// fp32 residual stream; bf16 WMMA GEMMs (128x128 block tile, 8 waves) and
// flash attention; fused bias/residual/GELU epilogues; one-time bf16 weight
// conversion when workspace allows.
// ---------------------------------------------------------------------------

typedef __attribute__((ext_vector_type(16))) __bf16 v16bf;
typedef __attribute__((ext_vector_type(8)))  float  v8f;

#define BB   2
#define SS   2048
#define PP   512
#define NL   1536        // latent length S-P
#define DD   1024
#define HH   16
#define DHD  64
#define IDD  1024        // H*DH
#define VV   20000
#define LL   4
#define FFD  4096
#define QSCALE 0.125f    // DH^-0.5

// ---- fragment gather: two contiguous 8-element (16B) LDS runs ----
__device__ inline v16bf ld_frag(const __bf16* p0, const __bf16* p1) {
  v16bf r;
#pragma unroll
  for (int t = 0; t < 8; ++t) { r[t] = p0[t]; r[8 + t] = p1[t]; }
  return r;
}

// ---------------------------------------------------------------------------
// Embedding: emb = token_emb[x] + pos_emb ; split into prefix / latent stream
// ---------------------------------------------------------------------------
__global__ void __launch_bounds__(256) embed_k(const int* __restrict__ x,
    const float* __restrict__ tok, const float* __restrict__ pos,
    float* __restrict__ prefix, float* __restrict__ lat)
{
  int row = blockIdx.x;              // b*S + s
  int b = row / SS, s = row % SS;
  int id = x[row];
  const float* tr = tok + (size_t)id * DD;
  const float* pr = pos + (size_t)s * DD;
  float* dst = (s < PP) ? (prefix + ((size_t)b * PP + s) * DD)
                        : (lat    + ((size_t)b * NL + (s - PP)) * DD);
  for (int d = threadIdx.x; d < DD; d += 256) dst[d] = tr[d] + pr[d];
}

// ---------------------------------------------------------------------------
// LayerNorm: fp32 row -> bf16 row (feeds WMMA GEMM A operand)
// ---------------------------------------------------------------------------
__global__ void __launch_bounds__(256) ln_k(const float* __restrict__ X,
    const float* __restrict__ sg, const float* __restrict__ bg,
    __bf16* __restrict__ Y, int Dd)
{
  int row = blockIdx.x, tid = threadIdx.x;
  const float* xr = X + (size_t)row * Dd;
  float s1 = 0.f, s2 = 0.f;
  for (int d = tid; d < Dd; d += 256) { float t = xr[d]; s1 += t; s2 += t * t; }
#pragma unroll
  for (int m = 16; m >= 1; m >>= 1) { s1 += __shfl_xor(s1, m, 32); s2 += __shfl_xor(s2, m, 32); }
  __shared__ float r1[8], r2[8];
  if ((tid & 31) == 0) { r1[tid >> 5] = s1; r2[tid >> 5] = s2; }
  __syncthreads();
  float t1 = 0.f, t2 = 0.f;
#pragma unroll
  for (int i = 0; i < 8; ++i) { t1 += r1[i]; t2 += r2[i]; }
  float mean = t1 / (float)Dd;
  float var  = t2 / (float)Dd - mean * mean;
  float rstd = rsqrtf(var + 1e-5f);
  __bf16* yr = Y + (size_t)row * Dd;
  for (int d = tid; d < Dd; d += 256)
    yr[d] = (__bf16)((xr[d] - mean) * rstd * sg[d] + bg[d]);
}

// ---------------------------------------------------------------------------
// WMMA GEMM: C[M,N] = A_bf16[M,K] * B[K,N] (+bias)(+res)(gelu)
// Block: 256 threads = 8 waves (2x4). Block tile 128x128, wave tile 64x32
// (4x2 v_wmma_f32_16x16x32_bf16), K-step 32.
// B comes from Bw16 (preconverted bf16) if non-null, else fp32 Bw32 converted
// during the LDS stage. M, K multiples of 128/32; N edge-guarded (V=20000).
// ---------------------------------------------------------------------------
__global__ void __launch_bounds__(256) gemm_k(const __bf16* __restrict__ A,
    const float* __restrict__ Bw32, const __bf16* __restrict__ Bw16,
    float* __restrict__ Cf, __bf16* __restrict__ Cb,
    const float* __restrict__ bias, const float* __restrict__ res,
    int M, int N, int K, int act)
{
  __shared__ __align__(16) __bf16 As[128][32];   // [m][k]  8KB
  __shared__ __align__(16) __bf16 Bs[128][32];   // [n][k]  8KB (transposed)
  int tid  = threadIdx.x;
  int lane = tid & 31, wid = tid >> 5;
  int wm = wid >> 2, wn = wid & 3;               // 2 x 4 wave grid
  int lm = lane & 15, lh = lane >> 4;
  int n0 = blockIdx.x * 128, m0 = blockIdx.y * 128;

  v8f acc[4][2];
#pragma unroll
  for (int a = 0; a < 4; ++a)
#pragma unroll
    for (int b = 0; b < 2; ++b)
#pragma unroll
      for (int t = 0; t < 8; ++t) acc[a][b][t] = 0.f;

  for (int kk = 0; kk < K; kk += 32) {
    __syncthreads();
    { // A stage: 128 rows x 32 bf16; thread t: row t>>1, 16-elem half (t&1)
      int r = tid >> 1, c = (tid & 1) * 16;
      const unsigned int* ag =
          reinterpret_cast<const unsigned int*>(A + (size_t)(m0 + r) * K + kk + c);
      unsigned int* as = reinterpret_cast<unsigned int*>(&As[r][c]);
#pragma unroll
      for (int t = 0; t < 8; ++t) as[t] = ag[t];
    }
    { // B stage: rows kk..kk+32, cols n0..n0+128 -> Bs[n][k]
      int k  = tid >> 3;            // 0..31
      int nc = (tid & 7) * 16;      // 0..112
      if (Bw16) {
        const __bf16* bg = Bw16 + (size_t)(kk + k) * N + n0 + nc;
        if (n0 + nc + 16 <= N) {
#pragma unroll
          for (int c = 0; c < 16; ++c) Bs[nc + c][k] = bg[c];
        } else {
#pragma unroll
          for (int c = 0; c < 16; ++c)
            Bs[nc + c][k] = (n0 + nc + c < N) ? bg[c] : (__bf16)0.f;
        }
        if (kk + 32 < K)
          __builtin_prefetch(Bw16 + (size_t)(kk + 32 + k) * N + n0 + nc, 0, 0);
      } else {
        const float* bg = Bw32 + (size_t)(kk + k) * N + n0 + nc;
#pragma unroll
        for (int c = 0; c < 16; ++c)
          Bs[nc + c][k] = (n0 + nc + c < N) ? (__bf16)bg[c] : (__bf16)0.f;
        if (kk + 32 < K)
          __builtin_prefetch(Bw32 + (size_t)(kk + 32 + k) * N + n0 + nc, 0, 0);
      }
    }
    __syncthreads();

    v16bf af[4], bfv[2];
#pragma unroll
    for (int mi = 0; mi < 4; ++mi) {
      const __bf16* rp = &As[wm * 64 + mi * 16 + lm][0];
      af[mi] = ld_frag(rp + lh * 8, rp + 16 + lh * 8);
    }
#pragma unroll
    for (int ni = 0; ni < 2; ++ni) {
      const __bf16* rp = &Bs[wn * 32 + ni * 16 + lm][0];
      bfv[ni] = ld_frag(rp + lh * 8, rp + 16 + lh * 8);
    }
#pragma unroll
    for (int mi = 0; mi < 4; ++mi)
#pragma unroll
      for (int ni = 0; ni < 2; ++ni)
        acc[mi][ni] = __builtin_amdgcn_wmma_f32_16x16x32_bf16(
            false, af[mi], false, bfv[ni], (short)0, acc[mi][ni], false, false);
  }

  // epilogue: element v of tile (mi,ni) -> (m = mi*16+v+8*lh, n = ni*16+lm)
#pragma unroll
  for (int mi = 0; mi < 4; ++mi)
#pragma unroll
    for (int ni = 0; ni < 2; ++ni)
#pragma unroll
      for (int v = 0; v < 8; ++v) {
        int m = m0 + wm * 64 + mi * 16 + v + 8 * lh;
        int n = n0 + wn * 32 + ni * 16 + lm;
        if (n < N) {
          float val = acc[mi][ni][v];
          if (bias) val += bias[n];
          if (res)  val += res[(size_t)m * N + n];
          if (act)  val = 0.5f * val * (1.f + erff(val * 0.70710678118f));
          size_t o = (size_t)m * N + n;
          if (Cf) Cf[o] = val;
          if (Cb) Cb[o] = (__bf16)val;
        }
      }
}

// ---------------------------------------------------------------------------
// Split heads + scale + RoPE(first 32 dims), fp32 [b*N, ld] -> bf16 [b,h,N,64]
// ---------------------------------------------------------------------------
__global__ void __launch_bounds__(256) head_prep_k(const float* __restrict__ src,
    int ld, int coloff, __bf16* __restrict__ dst, int Nrows, int posbase,
    float scale, int dorope)
{
  size_t total = (size_t)BB * HH * (size_t)Nrows * DHD;
  for (size_t idx = (size_t)blockIdx.x * 256 + threadIdx.x; idx < total;
       idx += (size_t)gridDim.x * 256) {
    int dh = (int)(idx & 63);
    size_t r = idx >> 6;
    int i = (int)(r % Nrows); r /= Nrows;
    int h = (int)(r % HH);    int b = (int)(r / HH);
    const float* base = src + ((size_t)b * Nrows + i) * ld + coloff + h * DHD;
    float val = base[dh];
    if (dorope && dh < 32) {
      int d = dh & 15;
      float inv = __powf(10000.f, -(float)d * (1.f / 16.f));
      float ang = (float)(posbase + i) * inv;
      float sn, cs; __sincosf(ang, &sn, &cs);
      float other = base[(dh < 16) ? dh + 16 : dh - 16];
      val = (dh < 16) ? (val * cs - other * sn) : (val * cs + other * sn);
    }
    dst[idx] = (__bf16)(val * scale);
  }
}

// ---------------------------------------------------------------------------
// Cross-attn K/V prep: concat(prefix KV, latent KV) -> head-major bf16 + RoPE(k)
// ---------------------------------------------------------------------------
__global__ void __launch_bounds__(256) kv_prep_ca_k(const float* __restrict__ kvc,
    const float* __restrict__ kvin, __bf16* __restrict__ kdst, __bf16* __restrict__ vdst)
{
  const size_t total = (size_t)BB * HH * SS * DHD;
  for (size_t idx = (size_t)blockIdx.x * 256 + threadIdx.x; idx < total;
       idx += (size_t)gridDim.x * 256) {
    int dh = (int)(idx & 63);
    size_t r = idx >> 6;
    int j = (int)(r % SS); r /= SS;
    int h = (int)(r % HH); int b = (int)(r / HH);
    const float* srow = (j < PP) ? (kvc  + ((size_t)b * PP + j) * (2 * IDD))
                                 : (kvin + ((size_t)b * NL + (j - PP)) * (2 * IDD));
    float kval = srow[h * DHD + dh];
    float vval = srow[IDD + h * DHD + dh];
    if (dh < 32) {
      int d = dh & 15;
      float inv = __powf(10000.f, -(float)d * (1.f / 16.f));
      float ang = (float)j * inv;
      float sn, cs; __sincosf(ang, &sn, &cs);
      float other = srow[h * DHD + ((dh < 16) ? dh + 16 : dh - 16)];
      kval = (dh < 16) ? (kval * cs - other * sn) : (kval * cs + other * sn);
    }
    kdst[idx] = (__bf16)kval;
    vdst[idx] = (__bf16)vval;
  }
}

// ---------------------------------------------------------------------------
// Flash attention, one wave per (b,h, 16-row q tile). Causal: j <= i + offs.
// QK^T and P*V both via v_wmma_f32_16x16x32_bf16; online softmax in registers.
// ---------------------------------------------------------------------------
__global__ void __launch_bounds__(32) attn_k(const __bf16* __restrict__ Q,
    const __bf16* __restrict__ Kh, const __bf16* __restrict__ Vh,
    __bf16* __restrict__ Out, int Nq, int Jt, int offs, int Hn)
{
  __shared__ __align__(16) __bf16 Qs[16][64];
  __shared__ __align__(16) __bf16 Kt[32][64];   // [kv][dim]
  __shared__ __align__(16) __bf16 Vt[64][32];   // [dim][kv] (transposed)
  __shared__ __align__(16) __bf16 Pt[16][32];   // probs, [m][kv]
  int lane = threadIdx.x;
  int lm = lane & 15, lh = lane >> 4;
  int qt = blockIdx.x, bh = blockIdx.y;
  int b = bh / Hn, h = bh % Hn;
  int qbase = qt * 16;
  const __bf16* qp = Q  + ((size_t)bh * Nq + qbase) * DHD;
  const __bf16* kp = Kh + (size_t)bh * Jt * DHD;
  const __bf16* vp = Vh + (size_t)bh * Jt * DHD;

  { // stage Q tile (16x64)
    int r = lane >> 1, off = (lane & 1) * 32;
    const unsigned int* s = reinterpret_cast<const unsigned int*>(qp + (size_t)r * DHD + off);
    unsigned int* d = reinterpret_cast<unsigned int*>(&Qs[r][off]);
#pragma unroll
    for (int t = 0; t < 16; ++t) d[t] = s[t];
  }
  __syncthreads();
  v16bf aq[2];
#pragma unroll
  for (int c = 0; c < 2; ++c)
    aq[c] = ld_frag(&Qs[lm][c * 32 + lh * 8], &Qs[lm][c * 32 + 16 + lh * 8]);

  float mrow[8], lrow[8];
  v8f oc[4];
#pragma unroll
  for (int v = 0; v < 8; ++v) { mrow[v] = -3.0e38f; lrow[v] = 0.f; }
#pragma unroll
  for (int jt = 0; jt < 4; ++jt)
#pragma unroll
    for (int t = 0; t < 8; ++t) oc[jt][t] = 0.f;

  int jmax = qbase + 15 + offs + 1; if (jmax > Jt) jmax = Jt;
  for (int kv = 0; kv < jmax; kv += 32) {
    __syncthreads();
    { // stage K rows [kv..kv+32) and V transposed
      const unsigned int* s = reinterpret_cast<const unsigned int*>(kp + (size_t)(kv + lane) * DHD);
      unsigned int* d = reinterpret_cast<unsigned int*>(&Kt[lane][0]);
#pragma unroll
      for (int t = 0; t < 32; ++t) d[t] = s[t];
      const __bf16* vs = vp + (size_t)(kv + lane) * DHD;
#pragma unroll
      for (int t = 0; t < 64; ++t) Vt[t][lane] = vs[t];
    }
    __syncthreads();

    v8f s0, s1;
#pragma unroll
    for (int t = 0; t < 8; ++t) { s0[t] = 0.f; s1[t] = 0.f; }
#pragma unroll
    for (int c = 0; c < 2; ++c) {
      v16bf bk0 = ld_frag(&Kt[lm][c * 32 + lh * 8],      &Kt[lm][c * 32 + 16 + lh * 8]);
      v16bf bk1 = ld_frag(&Kt[16 + lm][c * 32 + lh * 8], &Kt[16 + lm][c * 32 + 16 + lh * 8]);
      s0 = __builtin_amdgcn_wmma_f32_16x16x32_bf16(false, aq[c], false, bk0, (short)0, s0, false, false);
      s1 = __builtin_amdgcn_wmma_f32_16x16x32_bf16(false, aq[c], false, bk1, (short)0, s1, false, false);
    }

    // online softmax; a row's 16 scores live across the 16 lanes of a half-wave
#pragma unroll
    for (int v = 0; v < 8; ++v) {
      int i  = qbase + v + 8 * lh;
      int j0 = kv + lm, j1 = kv + 16 + lm;
      float x0 = (j0 <= i + offs) ? s0[v] : -3.0e38f;
      float x1 = (j1 <= i + offs) ? s1[v] : -3.0e38f;
      float mx = fmaxf(x0, x1);
#pragma unroll
      for (int mk = 1; mk < 16; mk <<= 1) mx = fmaxf(mx, __shfl_xor(mx, mk, 32));
      float newm = fmaxf(mrow[v], mx);
      float corr = __expf(mrow[v] - newm);
      float p0 = __expf(x0 - newm);
      float p1 = __expf(x1 - newm);
      float rs = p0 + p1;
#pragma unroll
      for (int mk = 1; mk < 16; mk <<= 1) rs += __shfl_xor(rs, mk, 32);
      lrow[v] = lrow[v] * corr + rs;
      mrow[v] = newm;
#pragma unroll
      for (int jt = 0; jt < 4; ++jt) oc[jt][v] *= corr;
      Pt[v + 8 * lh][lm]      = (__bf16)p0;
      Pt[v + 8 * lh][16 + lm] = (__bf16)p1;
    }
    __syncthreads();

    v16bf pa = ld_frag(&Pt[lm][lh * 8], &Pt[lm][16 + lh * 8]);
#pragma unroll
    for (int jt = 0; jt < 4; ++jt) {
      v16bf bv = ld_frag(&Vt[jt * 16 + lm][lh * 8], &Vt[jt * 16 + lm][16 + lh * 8]);
      oc[jt] = __builtin_amdgcn_wmma_f32_16x16x32_bf16(false, pa, false, bv, (short)0, oc[jt], false, false);
    }
  }

  // normalize + store merged-heads bf16: out[(b*Nq+i), h*64+dh]
#pragma unroll
  for (int v = 0; v < 8; ++v) {
    float inv = 1.0f / lrow[v];
    int i = qbase + v + 8 * lh;
    __bf16* orow = Out + ((size_t)b * Nq + i) * ((size_t)Hn * DHD) + h * DHD;
#pragma unroll
    for (int jt = 0; jt < 4; ++jt)
      orow[jt * 16 + lm] = (__bf16)(oc[jt][v] * inv);
  }
}

// ---------------------------------------------------------------------------
__global__ void __launch_bounds__(256) f2bf_k(const float* __restrict__ in,
    __bf16* __restrict__ out, size_t n)
{
  for (size_t i = (size_t)blockIdx.x * 256 + threadIdx.x; i < n;
       i += (size_t)gridDim.x * 256)
    out[i] = (__bf16)in[i];
}

// ---------------------------------------------------------------------------
static inline void launch_gemm(hipStream_t st, const __bf16* A,
                               const float* B32, const __bf16* B16,
                               float* Cf, __bf16* Cb, const float* bias,
                               const float* res, int M, int N, int K, int act)
{
  dim3 g((unsigned)((N + 127) / 128), (unsigned)(M / 128));
  gemm_k<<<g, 256, 0, st>>>(A, B32, B16, Cf, Cb, bias, res, M, N, K, act);
}

extern "C" void kernel_launch(void* const* d_in, const int* in_sizes, int n_in,
                              void* d_out, int out_size, void* d_ws, size_t ws_size,
                              hipStream_t stream)
{
  (void)in_sizes; (void)n_in; (void)out_size;
  const int*   x            = (const int*)  d_in[0];
  // d_in[1] = prefix_mask: all-true, no-op in the reference masking
  const float* token_emb    = (const float*)d_in[2];
  const float* pos_emb      = (const float*)d_in[3];
  const float* ca_norm_s    = (const float*)d_in[4];
  const float* ca_norm_b    = (const float*)d_in[5];
  const float* ca_ctx_norm_s= (const float*)d_in[6];
  const float* ca_ctx_norm_b= (const float*)d_in[7];
  const float* ca_wq        = (const float*)d_in[8];
  const float* ca_wkv       = (const float*)d_in[9];
  const float* ca_wo        = (const float*)d_in[10];
  const float* ca_bo        = (const float*)d_in[11];
  const float* ca_ff_ln_s   = (const float*)d_in[12];
  const float* ca_ff_ln_b   = (const float*)d_in[13];
  const float* ca_ff_w1     = (const float*)d_in[14];
  const float* ca_ff_w2     = (const float*)d_in[15];
  const float* l_norm_s     = (const float*)d_in[16];
  const float* l_norm_b     = (const float*)d_in[17];
  const float* l_wqkv       = (const float*)d_in[18];
  const float* l_wo         = (const float*)d_in[19];
  const float* l_ff_ln_s    = (const float*)d_in[20];
  const float* l_ff_ln_b    = (const float*)d_in[21];
  const float* l_ff_w1      = (const float*)d_in[22];
  const float* l_ff_w2      = (const float*)d_in[23];
  const float* w_logits     = (const float*)d_in[24];

  const int Mlat = BB * NL;   // 3072
  const int Mpre = BB * PP;   // 1024

  // ---- workspace carve: activations first, then (optional) bf16 weights ----
  char* w = (char*)d_ws;
  auto carve = [&](size_t bytes) -> void* {
    void* p = (void*)w;
    w += (bytes + 255) & ~(size_t)255;
    return p;
  };
  float*  lat    = (float*) carve((size_t)Mlat * DD * 4);
  float*  prefix = (float*) carve((size_t)Mpre * DD * 4);
  __bf16* xn     = (__bf16*)carve((size_t)Mlat * DD * 2);
  __bf16* cn     = (__bf16*)carve((size_t)Mpre * DD * 2);
  float*  bigf   = (float*) carve((size_t)Mlat * (3 * IDD) * 4); // qkv / q+kv_in
  float*  kvc    = (float*) carve((size_t)Mpre * (2 * IDD) * 4);
  __bf16* qb     = (__bf16*)carve((size_t)BB * HH * NL * DHD * 2);
  __bf16* kb     = (__bf16*)carve((size_t)BB * HH * SS * DHD * 2);
  __bf16* vb     = (__bf16*)carve((size_t)BB * HH * SS * DHD * 2);
  __bf16* ao     = (__bf16*)carve((size_t)Mlat * IDD * 2);
  __bf16* ffh    = (__bf16*)carve((size_t)Mlat * FFD * 2);

  float*  qf   = bigf;                         // [3072,1024]
  float*  kvin = bigf + (size_t)Mlat * IDD;    // [3072,2048]

  // bf16 weight cache (one conversion pass per launch; deterministic)
  const size_t wq_e   = (size_t)DD * IDD;
  const size_t wkv_e  = (size_t)DD * 2 * IDD;
  const size_t wo_e   = (size_t)IDD * DD;
  const size_t w1_e   = (size_t)DD * FFD;
  const size_t w2_e   = (size_t)FFD * DD;
  const size_t qkv_e  = (size_t)DD * 3 * IDD;
  const size_t wlog_e = (size_t)DD * VV;
  size_t wgt_elems = wq_e + wkv_e + wo_e + w1_e + w2_e +
                     LL * (qkv_e + wo_e + w1_e + w2_e) + wlog_e;
  size_t used = (size_t)(w - (char*)d_ws);
  bool use16 = ws_size >= used + wgt_elems * 2 + 16384;

  __bf16 *wq16 = nullptr, *wkv16 = nullptr, *wo16 = nullptr, *w116 = nullptr,
         *w216 = nullptr, *lqkv16 = nullptr, *lwo16 = nullptr, *lw116 = nullptr,
         *lw216 = nullptr, *wlog16 = nullptr;
  if (use16) {
    wq16   = (__bf16*)carve(wq_e * 2);
    wkv16  = (__bf16*)carve(wkv_e * 2);
    wo16   = (__bf16*)carve(wo_e * 2);
    w116   = (__bf16*)carve(w1_e * 2);
    w216   = (__bf16*)carve(w2_e * 2);
    lqkv16 = (__bf16*)carve(LL * qkv_e * 2);
    lwo16  = (__bf16*)carve(LL * wo_e * 2);
    lw116  = (__bf16*)carve(LL * w1_e * 2);
    lw216  = (__bf16*)carve(LL * w2_e * 2);
    wlog16 = (__bf16*)carve(wlog_e * 2);
    f2bf_k<<<2048, 256, 0, stream>>>(ca_wq,    wq16,   wq_e);
    f2bf_k<<<2048, 256, 0, stream>>>(ca_wkv,   wkv16,  wkv_e);
    f2bf_k<<<2048, 256, 0, stream>>>(ca_wo,    wo16,   wo_e);
    f2bf_k<<<2048, 256, 0, stream>>>(ca_ff_w1, w116,   w1_e);
    f2bf_k<<<2048, 256, 0, stream>>>(ca_ff_w2, w216,   w2_e);
    f2bf_k<<<4096, 256, 0, stream>>>(l_wqkv,   lqkv16, LL * qkv_e);
    f2bf_k<<<2048, 256, 0, stream>>>(l_wo,     lwo16,  LL * wo_e);
    f2bf_k<<<4096, 256, 0, stream>>>(l_ff_w1,  lw116,  LL * w1_e);
    f2bf_k<<<4096, 256, 0, stream>>>(l_ff_w2,  lw216,  LL * w2_e);
    f2bf_k<<<4096, 256, 0, stream>>>(w_logits, wlog16, wlog_e);
  }
  auto sub16 = [](const __bf16* p, size_t off) -> const __bf16* {
    return p ? p + off : nullptr;
  };

  // ---- embedding ----
  embed_k<<<BB * SS, 256, 0, stream>>>(x, token_emb, pos_emb, prefix, lat);

  // ---- cross-attention block ----
  ln_k<<<Mlat, 256, 0, stream>>>(lat,    ca_norm_s,     ca_norm_b,     xn, DD);
  ln_k<<<Mpre, 256, 0, stream>>>(prefix, ca_ctx_norm_s, ca_ctx_norm_b, cn, DD);
  launch_gemm(stream, xn, ca_wq,  wq16,  qf,   nullptr, nullptr, nullptr, Mlat, IDD,     DD, 0);
  launch_gemm(stream, xn, ca_wkv, wkv16, kvin, nullptr, nullptr, nullptr, Mlat, 2 * IDD, DD, 0);
  launch_gemm(stream, cn, ca_wkv, wkv16, kvc,  nullptr, nullptr, nullptr, Mpre, 2 * IDD, DD, 0);
  head_prep_k<<<2048, 256, 0, stream>>>(qf, IDD, 0, qb, NL, PP, QSCALE, 1);
  kv_prep_ca_k<<<2048, 256, 0, stream>>>(kvc, kvin, kb, vb);
  attn_k<<<dim3(NL / 16, BB * HH), 32, 0, stream>>>(qb, kb, vb, ao, NL, SS, PP, HH);
  launch_gemm(stream, ao, ca_wo, wo16, lat, nullptr, ca_bo, lat, Mlat, DD, IDD, 0);
  ln_k<<<Mlat, 256, 0, stream>>>(lat, ca_ff_ln_s, ca_ff_ln_b, xn, DD);
  launch_gemm(stream, xn,  ca_ff_w1, w116, nullptr, ffh, nullptr, nullptr, Mlat, FFD, DD, 1);
  launch_gemm(stream, ffh, ca_ff_w2, w216, lat, nullptr, nullptr, lat, Mlat, DD, FFD, 0);

  // ---- self-attention layers ----
  for (int li = 0; li < LL; ++li) {
    ln_k<<<Mlat, 256, 0, stream>>>(lat, l_norm_s + li * DD, l_norm_b + li * DD, xn, DD);
    launch_gemm(stream, xn, l_wqkv + (size_t)li * qkv_e, sub16(lqkv16, (size_t)li * qkv_e),
                bigf, nullptr, nullptr, nullptr, Mlat, 3 * IDD, DD, 0);
    head_prep_k<<<2048, 256, 0, stream>>>(bigf, 3 * IDD, 0,       qb, NL, PP, QSCALE, 1);
    head_prep_k<<<2048, 256, 0, stream>>>(bigf, 3 * IDD, IDD,     kb, NL, PP, 1.0f,   1);
    head_prep_k<<<2048, 256, 0, stream>>>(bigf, 3 * IDD, 2 * IDD, vb, NL, 0,  1.0f,   0);
    attn_k<<<dim3(NL / 16, BB * HH), 32, 0, stream>>>(qb, kb, vb, ao, NL, NL, 0, HH);
    launch_gemm(stream, ao, l_wo + (size_t)li * wo_e, sub16(lwo16, (size_t)li * wo_e),
                lat, nullptr, nullptr, lat, Mlat, DD, IDD, 0);
    ln_k<<<Mlat, 256, 0, stream>>>(lat, l_ff_ln_s + li * DD, l_ff_ln_b + li * DD, xn, DD);
    launch_gemm(stream, xn, l_ff_w1 + (size_t)li * w1_e, sub16(lw116, (size_t)li * w1_e),
                nullptr, ffh, nullptr, nullptr, Mlat, FFD, DD, 1);
    launch_gemm(stream, ffh, l_ff_w2 + (size_t)li * w2_e, sub16(lw216, (size_t)li * w2_e),
                lat, nullptr, nullptr, lat, Mlat, DD, FFD, 0);
  }

  // ---- logits ----
  f2bf_k<<<2048, 256, 0, stream>>>(lat, xn, (size_t)Mlat * DD);
  launch_gemm(stream, xn, w_logits, wlog16, (float*)d_out, nullptr, nullptr, nullptr,
              Mlat, VV, DD, 0);
}
